// RWKV6Layer_8323646620460
// MI455X (gfx1250) — matrix operand order repaired and checked by reading, not compiled, not run on hardware
//
#include <hip/hip_runtime.h>

// ---------------------------------------------------------------------------
// RWKV6 layer for gfx1250 (MI455X). bf16 WMMA GEMMs (128x64 tile, 4 wmma per
// wave per K-step, double-buffered LDS staging, one barrier per K-step);
// all GEMM inputs pre-converted to bf16; N=160 weight zero-padded to 192 so
// every GEMM takes the unguarded vectorized path. Recurrent scan software-
// pipelined with per-channel decay state in VGPRs. Compile-only target.
// ---------------------------------------------------------------------------

typedef __bf16 bf16_t;
typedef __attribute__((ext_vector_type(16))) __bf16 v16bf;
typedef __attribute__((ext_vector_type(8)))  __bf16 v8bf;
typedef __attribute__((ext_vector_type(8)))  float  v8f;

static constexpr int BB = 4, TT = 2048, DD = 512;
static constexpr int HH = 4, DK = 64, DV = 128;
static constexpr size_t MR = (size_t)BB * TT;     // 8192 token rows
static constexpr size_t SZ = MR * DD;             // 512-wide plane (elements)
static constexpr int XXP = 192;                   // xx plane padded from 160

__device__ __forceinline__ unsigned bfbits(float f) {
  unsigned u = __builtin_bit_cast(unsigned, f);
  return (u + 0x7FFFu + ((u >> 16) & 1u)) >> 16;  // round-to-nearest-even
}
__device__ __forceinline__ bf16_t f2bf(float f) {
  unsigned short s = (unsigned short)bfbits(f);
  return __builtin_bit_cast(bf16_t, s);
}
__device__ __forceinline__ float bf2f(bf16_t b) {
  unsigned u = ((unsigned)__builtin_bit_cast(unsigned short, b)) << 16;
  return __builtin_bit_cast(float, u);
}

// ACT: 0 none, 1 tanh, 2 exp(-exp(x)), 3 relu^2, 4 sigmoid
template <int ACT>
__device__ __forceinline__ float apply_act(float x) {
  if (ACT == 1) return tanhf(x);
  if (ACT == 2) return expf(-expf(x));
  if (ACT == 3) { float r = x > 0.0f ? x : 0.0f; return r * r; }
  if (ACT == 4) return 1.0f / (1.0f + expf(-x));
  return x;
}

// ---------------------------------------------------------------------------
// fp32 -> bf16 bulk conversion
// ---------------------------------------------------------------------------
__global__ __launch_bounds__(256)
void cvt_bf16_kernel(const float* __restrict__ in, bf16_t* __restrict__ out, size_t n) {
  size_t i = ((size_t)blockIdx.x * 256 + threadIdx.x) * 4;
  if (i + 3 < n) {
    float4 v = *reinterpret_cast<const float4*>(in + i);
    uint2 p;
    p.x = bfbits(v.x) | (bfbits(v.y) << 16);
    p.y = bfbits(v.z) | (bfbits(v.w) << 16);
    *reinterpret_cast<uint2*>(out + i) = p;
  } else {
    for (; i < n; ++i) out[i] = f2bf(in[i]);
  }
}

// fp32 (rows_in x cols) -> bf16 (rows_out x cols) with zero row padding
__global__ __launch_bounds__(256)
void cvt_pad_kernel(const float* __restrict__ in, bf16_t* __restrict__ out,
                    int rows_in, int rows_out, int cols) {
  size_t i = ((size_t)blockIdx.x * 256 + threadIdx.x) * 4;
  size_t n = (size_t)rows_out * cols;
  if (i >= n) return;
  int row = (int)(i / cols);
  uint2 p;
  if (row < rows_in) {
    float4 v = *reinterpret_cast<const float4*>(in + i);   // cols%4==0 keeps rows aligned
    p.x = bfbits(v.x) | (bfbits(v.y) << 16);
    p.y = bfbits(v.z) | (bfbits(v.w) << 16);
  } else {
    p.x = 0u; p.y = 0u;
  }
  *reinterpret_cast<uint2*>(out + i) = p;
}

// ---------------------------------------------------------------------------
// Tiled bf16 WMMA GEMM:  C[M,N] = act(A[M,K] * W[N,K]^T + bias[N])
// Block 256 threads = 8 waves; tile BM=128 x BN=64; each wave 32x32 (4 wmma).
// Double-buffered LDS staging, one barrier per K-step. OUTBF: store bf16.
// Requires M%128==0, N%64==0, K%32==0, 16B-aligned rows (lda/ldw%8==0).
// ---------------------------------------------------------------------------
template <int ACT, bool OUTBF>
__global__ __launch_bounds__(256)
void gemm_bf16(const bf16_t* __restrict__ A, int lda,
               const bf16_t* __restrict__ W, int ldw,
               const float* __restrict__ bias,
               void* __restrict__ Cv, int ldc,
               int Mdim, int Ndim, int Kdim) {
  __shared__ bf16_t As[2][128][40];   // 80B rows: 16B-aligned, 20-bank stride
  __shared__ bf16_t Bs[2][64][40];

  const int tid  = threadIdx.x;
  const int wid  = tid >> 5;
  const int lane = tid & 31;
  const int half = lane >> 4;
  const int l16  = lane & 15;
  const int wm   = wid & 3;        // wave row group: 32 rows each
  const int wn   = wid >> 2;       // wave col group: 32 cols each
  const int m0   = blockIdx.y * 128;
  const int n0   = blockIdx.x * 64;

  v8f c00 = {}, c01 = {}, c10 = {}, c11 = {};

  auto stage = [&](int k0, int buf) {
#pragma unroll
    for (int it = 0; it < 2; ++it) {
      int flat = it * 256 + tid;             // 0..511 chunks of 16B
      int rrow = flat >> 2, chunk = flat & 3;
      const uint4* src = reinterpret_cast<const uint4*>(
          A + (size_t)(m0 + rrow) * lda + k0);
      *reinterpret_cast<uint4*>(&As[buf][rrow][chunk * 8]) = src[chunk];
    }
    {
      int rrow = tid >> 2, chunk = tid & 3;
      const uint4* src = reinterpret_cast<const uint4*>(
          W + (size_t)(n0 + rrow) * ldw + k0);
      *reinterpret_cast<uint4*>(&Bs[buf][rrow][chunk * 8]) = src[chunk];
    }
  };

  const int kiters = Kdim >> 5;
  stage(0, 0);
  __syncthreads();

  for (int kt = 0; kt < kiters; ++kt) {
    const int cur = kt & 1;
    if (kt + 1 < kiters) stage((kt + 1) << 5, cur ^ 1);

    // ---- fragments (A 16x32 / B 32x16 bf16 lane layouts) ----
    v16bf a0, a1, b0, b1;
    {
      const bf16_t* ar0 = &As[cur][wm * 32 + l16][0];
      const bf16_t* ar1 = &As[cur][wm * 32 + 16 + l16][0];
      v8bf lo0 = *reinterpret_cast<const v8bf*>(ar0 + 8 * half);
      v8bf hi0 = *reinterpret_cast<const v8bf*>(ar0 + 16 + 8 * half);
      v8bf lo1 = *reinterpret_cast<const v8bf*>(ar1 + 8 * half);
      v8bf hi1 = *reinterpret_cast<const v8bf*>(ar1 + 16 + 8 * half);
      a0 = __builtin_shufflevector(lo0, hi0, 0,1,2,3,4,5,6,7,8,9,10,11,12,13,14,15);
      a1 = __builtin_shufflevector(lo1, hi1, 0,1,2,3,4,5,6,7,8,9,10,11,12,13,14,15);
      const bf16_t* br0 = &Bs[cur][wn * 32 + l16][0];
      const bf16_t* br1 = &Bs[cur][wn * 32 + 16 + l16][0];
      v8bf bl0 = *reinterpret_cast<const v8bf*>(br0 + 8 * half);
      v8bf bh0 = *reinterpret_cast<const v8bf*>(br0 + 16 + 8 * half);
      v8bf bl1 = *reinterpret_cast<const v8bf*>(br1 + 8 * half);
      v8bf bh1 = *reinterpret_cast<const v8bf*>(br1 + 16 + 8 * half);
      b0 = __builtin_shufflevector(bl0, bh0, 0,1,2,3,4,5,6,7,8,9,10,11,12,13,14,15);
      b1 = __builtin_shufflevector(bl1, bh1, 0,1,2,3,4,5,6,7,8,9,10,11,12,13,14,15);
    }
    c00 = __builtin_amdgcn_wmma_f32_16x16x32_bf16(false, a0, false, b0, (short)0, c00, false, false);
    c01 = __builtin_amdgcn_wmma_f32_16x16x32_bf16(false, a0, false, b1, (short)0, c01, false, false);
    c10 = __builtin_amdgcn_wmma_f32_16x16x32_bf16(false, a1, false, b0, (short)0, c10, false, false);
    c11 = __builtin_amdgcn_wmma_f32_16x16x32_bf16(false, a1, false, b1, (short)0, c11, false, false);
    __syncthreads();   // next buffer staged AND this buffer consumed
  }

  // ---- epilogue: C/D layout lane n = lane%16, VGPR i -> row i + 8*half ----
  const v8f* acc[4] = {&c00, &c01, &c10, &c11};
#pragma unroll
  for (int fa = 0; fa < 2; ++fa) {
#pragma unroll
    for (int fb = 0; fb < 2; ++fb) {
      const v8f& c = *acc[fa * 2 + fb];
      int gn = n0 + wn * 32 + fb * 16 + l16;
      float bv = bias ? bias[gn] : 0.0f;
#pragma unroll
      for (int i = 0; i < 8; ++i) {
        int gm = m0 + wm * 32 + fa * 16 + 8 * half + i;
        float v = apply_act<ACT>(c[i] + bv);
        if (OUTBF) ((bf16_t*)Cv)[(size_t)gm * ldc + gn] = f2bf(v);
        else       ((float*)Cv)[(size_t)gm * ldc + gn] = v;
      }
    }
  }
}

// ---------------------------------------------------------------------------
// LayerNorm helpers
// ---------------------------------------------------------------------------
__device__ __forceinline__ float block_sum256(float v, float* red) {
  int tid = threadIdx.x;
  red[tid] = v; __syncthreads();
  for (int s = 128; s > 0; s >>= 1) {
    if (tid < s) red[tid] += red[tid + s];
    __syncthreads();
  }
  float r = red[0]; __syncthreads();
  return r;
}

__global__ __launch_bounds__(256)
void ln01_kernel(const float* __restrict__ x,
                 const float* __restrict__ g0, const float* __restrict__ b0,
                 const float* __restrict__ g1, const float* __restrict__ b1,
                 float* __restrict__ resid, float* __restrict__ h) {
  __shared__ float red[256];
  size_t row = blockIdx.x;
  int t = threadIdx.x;
  const float* xr = x + row * DD;
  float v0 = xr[t], v1 = xr[t + 256];

  float mean = block_sum256(v0 + v1, red) * (1.0f / DD);
  float d0 = v0 - mean, d1 = v1 - mean;
  float var = block_sum256(d0 * d0 + d1 * d1, red) * (1.0f / DD);
  float inv = rsqrtf(var + 1e-5f);
  float r0 = d0 * inv * g0[t] + b0[t];
  float r1 = d1 * inv * g0[t + 256] + b0[t + 256];
  resid[row * DD + t] = r0;
  resid[row * DD + t + 256] = r1;

  float mean2 = block_sum256(r0 + r1, red) * (1.0f / DD);
  float e0 = r0 - mean2, e1 = r1 - mean2;
  float var2 = block_sum256(e0 * e0 + e1 * e1, red) * (1.0f / DD);
  float inv2 = rsqrtf(var2 + 1e-5f);
  h[row * DD + t]       = e0 * inv2 * g1[t] + b1[t];
  h[row * DD + t + 256] = e1 * inv2 * g1[t + 256] + b1[t + 256];
}

__global__ __launch_bounds__(256)
void ln_kernel(const float* __restrict__ x,
               const float* __restrict__ g, const float* __restrict__ b,
               float* __restrict__ out) {
  __shared__ float red[256];
  size_t row = blockIdx.x;
  int t = threadIdx.x;
  const float* xr = x + row * DD;
  float v0 = xr[t], v1 = xr[t + 256];
  float mean = block_sum256(v0 + v1, red) * (1.0f / DD);
  float d0 = v0 - mean, d1 = v1 - mean;
  float var = block_sum256(d0 * d0 + d1 * d1, red) * (1.0f / DD);
  float inv = rsqrtf(var + 1e-5f);
  out[row * DD + t]       = d0 * inv * g[t] + b[t];
  out[row * DD + t + 256] = d1 * inv * g[t + 256] + b[t + 256];
}

// ---------------------------------------------------------------------------
// Token shift + mu-mix:  delta = shift(h)-h ; u0 = bf16(h + delta*mu)
// ---------------------------------------------------------------------------
__global__ __launch_bounds__(256)
void delta_u0_kernel(const float* __restrict__ h, const float* __restrict__ mu,
                     float* __restrict__ delta, bf16_t* __restrict__ u0) {
  size_t idx = (size_t)blockIdx.x * 256 + threadIdx.x;
  if (idx >= SZ) return;
  size_t row = idx >> 9;
  int col = (int)(idx & 511);
  int t = (int)(row & (TT - 1));
  float hv = h[idx];
  float pv = (t > 0) ? h[idx - DD] : 0.0f;
  float d = pv - hv;
  delta[idx] = d;
  u0[idx] = f2bf(hv + d * mu[col]);
}

// In-place on bf16: u[n] (currently mods m_n) -> bf16(h + delta * m_n)
__global__ __launch_bounds__(256)
void mods_to_u_kernel(const float* __restrict__ h, const float* __restrict__ delta,
                      bf16_t* __restrict__ u) {
  size_t idx = (size_t)blockIdx.x * 256 + threadIdx.x;
  if (idx >= SZ) return;
  float hv = h[idx], d = delta[idx];
#pragma unroll
  for (int n = 0; n < 5; ++n) {
    size_t p = (size_t)n * SZ + idx;
    u[p] = f2bf(hv + d * bf2f(u[p]));
  }
}

// FFN token shift: ufk = bf16(h2 + d2*fk_mu) ; ufr = bf16(h2 + d2*fr_mu)
__global__ __launch_bounds__(256)
void delta_ffn_kernel(const float* __restrict__ h2,
                      const float* __restrict__ fk_mu, const float* __restrict__ fr_mu,
                      bf16_t* __restrict__ ufk, bf16_t* __restrict__ ufr) {
  size_t idx = (size_t)blockIdx.x * 256 + threadIdx.x;
  if (idx >= SZ) return;
  size_t row = idx >> 9;
  int col = (int)(idx & 511);
  int t = (int)(row & (TT - 1));
  float hv = h2[idx];
  float pv = (t > 0) ? h2[idx - DD] : 0.0f;
  float d = pv - hv;
  ufk[idx] = f2bf(hv + d * fk_mu[col]);
  ufr[idx] = f2bf(hv + d * fr_mu[col]);
}

// ---------------------------------------------------------------------------
// Recurrent scan: one block per (b,h); thread j owns v-column j (DV=128).
// S[DK] in VGPRs; r/k/dec broadcast via LDS; next step prefetched into regs.
// ---------------------------------------------------------------------------
__global__ __launch_bounds__(128)
void scan_kernel(const float* __restrict__ r, const float* __restrict__ k,
                 const float* __restrict__ dec, const float* __restrict__ v,
                 const float* __restrict__ bonus, float* __restrict__ o) {
  __shared__ float sr[DK], sk[DK], sd[DK], sb[DK];
  int b  = blockIdx.x >> 2;
  int hh = blockIdx.x & 3;
  int j  = threadIdx.x;                  // 0..127
  if (j < DK) sb[j] = bonus[hh * DK + j];
  float S[DK];
#pragma unroll
  for (int i = 0; i < DK; ++i) S[i] = 0.0f;

  size_t rbase = (size_t)b * TT * (HH * DK) + hh * DK + j;
  size_t vbase = (size_t)b * TT * (HH * DV) + hh * DV + j;
  float cr = 0.0f, ck = 0.0f, cd = 0.0f;
  if (j < DK) { cr = r[rbase]; ck = k[rbase]; cd = dec[rbase]; }
  float cv = v[vbase];
  __syncthreads();

  for (int t = 0; t < TT; ++t) {
    if (j < DK) { sr[j] = cr; sk[j] = ck; sd[j] = cd; }
    float vj = cv;
    __syncthreads();

    // software-pipeline: issue next step's loads before the compute chain
    float nr = 0.0f, nk = 0.0f, nd = 0.0f, nv = 0.0f;
    if (t + 1 < TT) {
      size_t rb = rbase + (size_t)(t + 1) * (HH * DK);
      if (j < DK) { nr = r[rb]; nk = k[rb]; nd = dec[rb]; }
      nv = v[vbase + (size_t)(t + 1) * (HH * DV)];
    }

    float acc = 0.0f;
#pragma unroll
    for (int i = 0; i < DK; ++i) {
      float kv = sk[i] * vj;
      acc += sr[i] * (S[i] + sb[i] * kv);
      S[i] = sd[i] * S[i] + kv;
    }
    o[vbase + (size_t)t * (HH * DV)] = acc;
    __syncthreads();
    cr = nr; ck = nk; cd = nd; cv = nv;
  }
}

// ---------------------------------------------------------------------------
// Per-head GroupNorm + swish gate; reads fp32 o, writes bf16 for o_w GEMM.
// ---------------------------------------------------------------------------
__global__ __launch_bounds__(128)
void gn_gate_kernel(const float* __restrict__ o, const float* __restrict__ g,
                    const float* __restrict__ gn_g, const float* __restrict__ gn_b,
                    bf16_t* __restrict__ oo) {
  __shared__ float red[128];
  size_t row = blockIdx.x >> 2;
  int hh = blockIdx.x & 3;
  int j = threadIdx.x;
  size_t base = row * DD + (size_t)hh * DV;
  float v = o[base + j];

  red[j] = v; __syncthreads();
  for (int s = 64; s > 0; s >>= 1) { if (j < s) red[j] += red[j + s]; __syncthreads(); }
  float mean = red[0] * (1.0f / DV); __syncthreads();
  float d = v - mean;
  red[j] = d * d; __syncthreads();
  for (int s = 64; s > 0; s >>= 1) { if (j < s) red[j] += red[j + s]; __syncthreads(); }
  float inv = rsqrtf(red[0] * (1.0f / DV) + 1e-5f); __syncthreads();

  float gv = g[base + j];
  float sw = gv / (1.0f + expf(-gv));            // g * sigmoid(g)
  oo[base + j] = f2bf((d * inv * gn_g[j] + gn_b[j]) * sw);
}

__global__ __launch_bounds__(256)
void add_kernel(float* __restrict__ a, const float* __restrict__ resid) {
  size_t idx = (size_t)blockIdx.x * 256 + threadIdx.x;
  if (idx < SZ) a[idx] += resid[idx];
}

__global__ __launch_bounds__(256)
void final_kernel(const float* __restrict__ z, const float* __restrict__ rr,
                  const float* __restrict__ yv, float* __restrict__ out) {
  size_t idx = (size_t)blockIdx.x * 256 + threadIdx.x;
  if (idx < SZ) out[idx] = z[idx] + rr[idx] * yv[idx];
}

// ---------------------------------------------------------------------------
// Launch
// ---------------------------------------------------------------------------
extern "C" void kernel_launch(void* const* d_in, const int* in_sizes, int n_in,
                              void* d_out, int out_size, void* d_ws, size_t ws_size,
                              hipStream_t stream) {
  const float* x     = (const float*)d_in[0];
  const float* ln0_g = (const float*)d_in[1];
  const float* ln0_b = (const float*)d_in[2];
  const float* ln1_g = (const float*)d_in[3];
  const float* ln1_b = (const float*)d_in[4];
  const float* ln2_g = (const float*)d_in[5];
  const float* ln2_b = (const float*)d_in[6];
  const float* xp_mu = (const float*)d_in[7];
  const float* xp_w1 = (const float*)d_in[8];   // (160, 512)
  const float* xp_w2 = (const float*)d_in[9];   // (512, 160)
  const float* x_bias= (const float*)d_in[10];  // (5, 512)
  const float* r_w   = (const float*)d_in[11];  // (256, 512)
  const float* w_A   = (const float*)d_in[12];  // (64, 512)
  const float* w_B   = (const float*)d_in[13];  // (256, 64)
  const float* w_b   = (const float*)d_in[14];  // (256,)
  const float* k_w   = (const float*)d_in[15];  // (256, 512)
  const float* v_w   = (const float*)d_in[16];  // (512, 512)
  const float* g_w   = (const float*)d_in[17];  // (512, 512)
  const float* bonus = (const float*)d_in[18];  // (4, 64)
  const float* gn_g  = (const float*)d_in[19];  // (128,)
  const float* gn_b  = (const float*)d_in[20];
  const float* o_w   = (const float*)d_in[21];  // (512, 512)
  const float* fk_mu = (const float*)d_in[22];
  const float* fk_w  = (const float*)d_in[23];  // (1024, 512)
  const float* fr_mu = (const float*)d_in[24];
  const float* fr_w  = (const float*)d_in[25];  // (512, 512)
  const float* fv_w  = (const float*)d_in[26];  // (512, 1024)
  float* out = (float*)d_out;

  // ---- workspace carve-out (all section sizes are 16B multiples) ----
  char* cur = (char*)d_ws;
  auto allocF = [&](size_t n) { float* p = (float*)cur; cur += n * sizeof(float); return p; };
  auto allocB = [&](size_t n) { bf16_t* p = (bf16_t*)cur; cur += n * sizeof(bf16_t); return p; };

  float*  resid = allocF(SZ);
  float*  hbuf  = allocF(SZ);          // h, later h2
  float*  delta = allocF(SZ);
  float*  rbuf  = allocF(MR * 256);
  float*  decb  = allocF(MR * 256);
  float*  kbuf  = allocF(MR * 256);
  float*  vbuf  = allocF(SZ);          // v, later yv
  float*  gbuf  = allocF(SZ);
  float*  obuf  = allocF(SZ);          // scan output
  float*  abuf  = allocF(SZ);          // a -> z in place
  float*  rrbuf = allocF(SZ);          // sigmoid gate (fp32)

  bf16_t* u0bf  = allocB(SZ);          // mu-mixed h; reused as ufk
  bf16_t* xxbf  = allocB(MR * XXP);    // padded xx plane (cols 160..191 unused)
  bf16_t* ubf   = allocB(5 * SZ);      // mods -> mixed inputs; [0] reused as ufr,
                                       // [1..2] reused as kk (MR*1024)
  bf16_t* wtbf  = allocB(MR * 64);
  bf16_t* oobf  = allocB(SZ);

  bf16_t* xp_w1b = allocB((size_t)XXP * 512);   // zero-padded 192x512
  bf16_t* xp_w2b = allocB(512 * 160);
  bf16_t* r_wb   = allocB(256 * 512);
  bf16_t* w_Ab   = allocB(64 * 512);
  bf16_t* w_Bb   = allocB(256 * 64);
  bf16_t* k_wb   = allocB(256 * 512);
  bf16_t* v_wb   = allocB(512 * 512);
  bf16_t* g_wb   = allocB(512 * 512);
  bf16_t* o_wb   = allocB(512 * 512);
  bf16_t* fk_wb  = allocB(1024 * 512);
  bf16_t* fr_wb  = allocB(512 * 512);
  bf16_t* fv_wb  = allocB(512 * 1024);

  bf16_t* kkbf  = ubf + SZ;            // MR*1024 bf16 spans ubf[1..2]
  bf16_t* ufkbf = u0bf;
  bf16_t* ufrbf = ubf;

  const int EW = (int)((SZ + 255) / 256);
  dim3 blk256(256), blk128(128);
  auto cvt = [&](const float* src, bf16_t* dst, size_t n) {
    cvt_bf16_kernel<<<(unsigned)((n / 4 + 255) / 256), blk256, 0, stream>>>(src, dst, n);
  };

  // 0) weights -> bf16 (xp_w1 zero-padded to 192 rows)
  cvt_pad_kernel<<<(unsigned)(((size_t)XXP * 512 / 4 + 255) / 256), blk256, 0, stream>>>(
      xp_w1, xp_w1b, 160, XXP, 512);
  cvt(xp_w2, xp_w2b, 512 * 160);
  cvt(r_w,  r_wb, 256 * 512);     cvt(w_A, w_Ab, 64 * 512);
  cvt(w_B,  w_Bb, 256 * 64);      cvt(k_w, k_wb, 256 * 512);
  cvt(v_w,  v_wb, 512 * 512);     cvt(g_w, g_wb, 512 * 512);
  cvt(o_w,  o_wb, 512 * 512);     cvt(fk_w, fk_wb, 1024 * 512);
  cvt(fr_w, fr_wb, 512 * 512);    cvt(fv_w, fv_wb, 512 * 1024);

  // 1) residual = ln0(x); h = ln1(residual)
  ln01_kernel<<<(unsigned)MR, blk256, 0, stream>>>(x, ln0_g, ln0_b, ln1_g, ln1_b, resid, hbuf);

  // 2) token shift + mu mix (bf16 out)
  delta_u0_kernel<<<EW, blk256, 0, stream>>>(hbuf, xp_mu, delta, u0bf);

  // 3) xx = tanh(u0 @ xp_w1^T), padded to 192 cols (160..191 compute tanh(0))
  gemm_bf16<1, true><<<dim3(3, 64), blk256, 0, stream>>>(
      u0bf, DD, xp_w1b, DD, nullptr, xxbf, XXP, (int)MR, XXP, DD);

  // 4) mods_n = xx[:, n*32:] @ xp_w2[:, n*32:]^T + x_bias[n]  (bf16 out)
  for (int n = 0; n < 5; ++n) {
    gemm_bf16<0, true><<<dim3(8, 64), blk256, 0, stream>>>(
        xxbf + n * 32, XXP, xp_w2b + n * 32, 160, x_bias + (size_t)n * DD,
        ubf + (size_t)n * SZ, DD, (int)MR, DD, 32);
  }

  // 5) u_n = h + delta * mods_n (in place, bf16)
  mods_to_u_kernel<<<EW, blk256, 0, stream>>>(hbuf, delta, ubf);

  // 6-11) projections
  gemm_bf16<0, false><<<dim3(4, 64), blk256, 0, stream>>>(
      ubf + 0 * SZ, DD, r_wb, DD, nullptr, rbuf, 256, (int)MR, 256, DD);
  gemm_bf16<1, true><<<dim3(1, 64), blk256, 0, stream>>>(
      ubf + 1 * SZ, DD, w_Ab, DD, nullptr, wtbf, 64, (int)MR, 64, DD);
  gemm_bf16<2, false><<<dim3(4, 64), blk256, 0, stream>>>(
      wtbf, 64, w_Bb, 64, w_b, decb, 256, (int)MR, 256, 64);
  gemm_bf16<0, false><<<dim3(4, 64), blk256, 0, stream>>>(
      ubf + 2 * SZ, DD, k_wb, DD, nullptr, kbuf, 256, (int)MR, 256, DD);
  gemm_bf16<0, false><<<dim3(8, 64), blk256, 0, stream>>>(
      ubf + 3 * SZ, DD, v_wb, DD, nullptr, vbuf, DD, (int)MR, DD, DD);
  gemm_bf16<0, false><<<dim3(8, 64), blk256, 0, stream>>>(
      ubf + 4 * SZ, DD, g_wb, DD, nullptr, gbuf, DD, (int)MR, DD, DD);

  // 12) recurrent scan
  scan_kernel<<<BB * HH, blk128, 0, stream>>>(rbuf, kbuf, decb, vbuf, bonus, obuf);

  // 13) per-head groupnorm + swish gate (bf16 out for o_w GEMM)
  gn_gate_kernel<<<(unsigned)(MR * HH), blk128, 0, stream>>>(obuf, gbuf, gn_g, gn_b, oobf);

  // 14) a = oo @ o_w^T ; 15) z = a + resid
  gemm_bf16<0, false><<<dim3(8, 64), blk256, 0, stream>>>(
      oobf, DD, o_wb, DD, nullptr, abuf, DD, (int)MR, DD, DD);
  add_kernel<<<EW, blk256, 0, stream>>>(abuf, resid);

  // 16) h2 = ln2(z) ; 17) FFN token shift mixes (bf16 out)
  ln_kernel<<<(unsigned)MR, blk256, 0, stream>>>(abuf, ln2_g, ln2_b, hbuf);
  delta_ffn_kernel<<<EW, blk256, 0, stream>>>(hbuf, fk_mu, fr_mu, ufkbf, ufrbf);

  // 18) kk = relu(ufk @ fk_w^T)^2 (bf16) ; 19) rr = sigmoid(ufr @ fr_w^T) (fp32)
  gemm_bf16<3, true><<<dim3(16, 64), blk256, 0, stream>>>(
      ufkbf, DD, fk_wb, DD, nullptr, kkbf, 1024, (int)MR, 1024, DD);
  gemm_bf16<4, false><<<dim3(8, 64), blk256, 0, stream>>>(
      ufrbf, DD, fr_wb, DD, nullptr, rrbuf, DD, (int)MR, DD, DD);

  // 20) yv = kk @ fv_w^T (K=1024)
  gemm_bf16<0, false><<<dim3(8, 64), blk256, 0, stream>>>(
      kkbf, 1024, fv_wb, 1024, nullptr, vbuf, DD, (int)MR, DD, 1024);

  // 21) out = z + rr * yv
  final_kernel<<<EW, blk256, 0, stream>>>(abuf, rrbuf, vbuf, out);
}
